// Decoder_14018773254638
// MI455X (gfx1250) — matrix operand order repaired and checked by reading, not compiled, not run on hardware
//
#include <hip/hip_runtime.h>

// ---------------- problem constants (from reference) ----------------
constexpr int kB = 64;     // batch
constexpr int kS = 128;    // source length
constexpr int kT = 32;     // target length
constexpr int kH = 1024;   // hidden
constexpr int kE = 512;    // embedding dim
constexpr int kV = 32000;  // vocab

typedef __attribute__((ext_vector_type(16))) __bf16 v16bf;
typedef __attribute__((ext_vector_type(8)))  __bf16 v8bf;
typedef __attribute__((ext_vector_type(8)))  float  v8f;

// ---------------- fp32 -> bf16 convert ----------------
__global__ void cvt_bf16_kernel(const float* __restrict__ in, __bf16* __restrict__ out, size_t n) {
    size_t i = (size_t)blockIdx.x * blockDim.x + threadIdx.x;
    if (i < n) out[i] = (__bf16)in[i];
}

// ---------------- generic WMMA GEMM:  C[M,N] = A[M,K] * W[N,K]^T + bias[N] ----------------
// A, W row-major bf16; C fp32 with row stride ldc. M multiple of 64, N multiple of 64, K mult of 32.
// Register blocking: each wave owns ONE 16-wide N tile and FOUR 16-tall M tiles, so the W
// operand is loaded once per k-step and shared by 4 independent WMMAs (loads/wmma = 2.5,
// W L2 traffic cut 4x vs one-tile-per-wave). 4 waves per block take adjacent N tiles.
__global__ void __launch_bounds__(128) wmma_gemm_kernel(
    const __bf16* __restrict__ A, int lda,
    const __bf16* __restrict__ W, int ldw,
    const float*  __restrict__ bias,
    float* __restrict__ C, long long ldc,
    int K, int N)
{
    const int lane = threadIdx.x & 31;
    const int wid  = threadIdx.x >> 5;           // 0..3
    const int n0   = (blockIdx.x * 4 + wid) * 16;
    if (n0 >= N) return;                          // whole wave exits together (EXEC all-1 for wmma)
    const int mb   = blockIdx.y * 64;             // 4 m-tiles: mb, mb+16, mb+32, mb+48
    const int r    = lane & 15;
    const int hi   = lane >> 4;                   // half-wave select

    const __bf16* wrow = W + (size_t)(n0 + r) * ldw;   // B operand: row of W = column n0+r of B
    const __bf16* arow0 = A + (size_t)(mb +  0 + r) * lda;
    const __bf16* arow1 = A + (size_t)(mb + 16 + r) * lda;
    const __bf16* arow2 = A + (size_t)(mb + 32 + r) * lda;
    const __bf16* arow3 = A + (size_t)(mb + 48 + r) * lda;

    v8f acc0 = {}, acc1 = {}, acc2 = {}, acc3 = {};

    for (int kb = 0; kb < K; kb += 32) {
        // B 32x16 layout (matches A): lanes 0-15 hold K 0..7 & 16..23; lanes 16-31 hold K 8..15 & 24..31
        union { v16bf v; v8bf h[2]; } bvu;
        bvu.h[0] = *reinterpret_cast<const v8bf*>(wrow + kb + 8 * hi);
        bvu.h[1] = *reinterpret_cast<const v8bf*>(wrow + kb + 16 + 8 * hi);
        v16bf bv = bvu.v;

        // ISA 16-bit A 16x32 layout: lanes 0-15 hold K 0..7 & 16..23; lanes 16-31 hold K 8..15 & 24..31
        union { v16bf v; v8bf h[2]; } a0, a1, a2, a3;
        a0.h[0] = *reinterpret_cast<const v8bf*>(arow0 + kb + 8 * hi);
        a0.h[1] = *reinterpret_cast<const v8bf*>(arow0 + kb + 16 + 8 * hi);
        a1.h[0] = *reinterpret_cast<const v8bf*>(arow1 + kb + 8 * hi);
        a1.h[1] = *reinterpret_cast<const v8bf*>(arow1 + kb + 16 + 8 * hi);
        a2.h[0] = *reinterpret_cast<const v8bf*>(arow2 + kb + 8 * hi);
        a2.h[1] = *reinterpret_cast<const v8bf*>(arow2 + kb + 16 + 8 * hi);
        a3.h[0] = *reinterpret_cast<const v8bf*>(arow3 + kb + 8 * hi);
        a3.h[1] = *reinterpret_cast<const v8bf*>(arow3 + kb + 16 + 8 * hi);

        // 4 independent WMMAs sharing the B operand
        acc0 = __builtin_amdgcn_wmma_f32_16x16x32_bf16(false, a0.v, false, bv, (short)0, acc0, false, false);
        acc1 = __builtin_amdgcn_wmma_f32_16x16x32_bf16(false, a1.v, false, bv, (short)0, acc1, false, false);
        acc2 = __builtin_amdgcn_wmma_f32_16x16x32_bf16(false, a2.v, false, bv, (short)0, acc2, false, false);
        acc3 = __builtin_amdgcn_wmma_f32_16x16x32_bf16(false, a3.v, false, bv, (short)0, acc3, false, false);
    }

    // C/D 32-bit 16x16 layout: VGPR e -> (M = e + 8*hi, N = lane&15)
    const int col = n0 + r;
    const float bi = bias ? bias[col] : 0.0f;
    #pragma unroll
    for (int e = 0; e < 8; ++e) {
        const int rowe = mb + hi * 8 + e;
        C[(size_t)(rowe +  0) * ldc + col] = acc0[e] + bi;
        C[(size_t)(rowe + 16) * ldc + col] = acc1[e] + bi;
        C[(size_t)(rowe + 32) * ldc + col] = acc2[e] + bi;
        C[(size_t)(rowe + 48) * ldc + col] = acc3[e] + bi;
    }
}

// ---------------- attention: scores -> softmax -> context (fp32) ----------------
__global__ void attn_kernel(const float* __restrict__ q,     // B x H
                            const float* __restrict__ keys,  // B x S x H
                            const float* __restrict__ enc,   // B x S x H
                            const float* __restrict__ Wa,    // H
                            const float* __restrict__ ba,    // 1
                            float* __restrict__ context)     // B x H
{
    const int b    = blockIdx.x;
    const int tid  = threadIdx.x;          // 256 threads = 8 waves
    const int lane = tid & 31;
    const int wave = tid >> 5;
    __shared__ float sw[kS];

    // phase 1: scores[s] = Wa . tanh(q[b] + keys[b,s]) + ba
    for (int s = wave; s < kS; s += 8) {
        const float* kr = keys + ((size_t)b * kS + s) * kH;
        const float* qr = q + (size_t)b * kH;
        float part = 0.0f;
        for (int h = lane; h < kH; h += 32)
            part += tanhf(qr[h] + kr[h]) * Wa[h];
        #pragma unroll
        for (int off = 16; off; off >>= 1) part += __shfl_xor(part, off, 32);
        if (lane == 0) sw[s] = part + ba[0];
    }
    __syncthreads();

    // phase 2: softmax over S=128 done by wave 0
    if (wave == 0) {
        float v[4], m = -1e30f;
        #pragma unroll
        for (int i = 0; i < 4; ++i) { v[i] = sw[lane + 32 * i]; m = fmaxf(m, v[i]); }
        #pragma unroll
        for (int off = 16; off; off >>= 1) m = fmaxf(m, __shfl_xor(m, off, 32));
        float ssum = 0.0f;
        #pragma unroll
        for (int i = 0; i < 4; ++i) { v[i] = __expf(v[i] - m); ssum += v[i]; }
        #pragma unroll
        for (int off = 16; off; off >>= 1) ssum += __shfl_xor(ssum, off, 32);
        const float inv = 1.0f / ssum;
        #pragma unroll
        for (int i = 0; i < 4; ++i) sw[lane + 32 * i] = v[i] * inv;
    }
    __syncthreads();

    // phase 3: context[h] = sum_s w[s] * enc[b,s,h]
    for (int h = tid; h < kH; h += 256) {
        const float* er = enc + (size_t)b * kS * kH + h;
        float acc = 0.0f;
        for (int s = 0; s < kS; ++s) acc += sw[s] * er[(size_t)s * kH];
        context[(size_t)b * kH + h] = acc;
    }
}

// ---------------- x = concat(embedding[target[:,t]], context) in bf16 ----------------
__global__ void build_x_kernel(const __bf16* __restrict__ emb,   // V x E
                               const int* __restrict__ target,   // B x T
                               int t,
                               const float* __restrict__ context,// B x H
                               __bf16* __restrict__ x)           // B x (E+H)
{
    constexpr int XW = kE + kH;
    const int i = blockIdx.x * blockDim.x + threadIdx.x;
    if (i >= kB * XW) return;
    const int b = i / XW, c = i % XW;
    if (c < kE) {
        const int tok = target[b * kT + t];
        x[(size_t)b * XW + c] = emb[(size_t)tok * kE + c];
    } else {
        x[(size_t)b * XW + c] = (__bf16)context[(size_t)b * kH + (c - kE)];
    }
}

// ---------------- GRU gates, in-place state update ----------------
__global__ void gru_gate_kernel(const float* __restrict__ gi,  // B x 3H
                                const float* __restrict__ gh,  // B x 3H
                                float* __restrict__ state,     // B x H (in/out)
                                __bf16* __restrict__ state_bf) // B x H (out)
{
    const int i = blockIdx.x * blockDim.x + threadIdx.x;
    if (i >= kB * kH) return;
    const int b = i / kH, h = i % kH;
    const size_t row = (size_t)b * 3 * kH;
    const float ir = gi[row + h], iz = gi[row + kH + h], in_ = gi[row + 2 * kH + h];
    const float hr = gh[row + h], hz = gh[row + kH + h], hn  = gh[row + 2 * kH + h];
    const float rg = 1.0f / (1.0f + __expf(-(ir + hr)));
    const float zg = 1.0f / (1.0f + __expf(-(iz + hz)));
    const float ng = tanhf(in_ + rg * hn);
    const float ns = (1.0f - zg) * ng + zg * state[i];
    state[i] = ns;
    state_bf[i] = (__bf16)ns;
}

// ---------------- state0 = encodings[:, -1, :] ----------------
__global__ void init_state_kernel(const float* __restrict__ enc,
                                  float* __restrict__ state, __bf16* __restrict__ state_bf)
{
    const int i = blockIdx.x * blockDim.x + threadIdx.x;
    if (i >= kB * kH) return;
    const int b = i / kH, h = i % kH;
    const float v = enc[((size_t)b * kS + (kS - 1)) * kH + h];
    state[i] = v;
    state_bf[i] = (__bf16)v;
}

// ---------------- out[:,0,:] = one-hot(1) ----------------
__global__ void sos_kernel(float* __restrict__ out)
{
    const int i = blockIdx.x * blockDim.x + threadIdx.x;
    if (i >= kB * kV) return;
    const int b = i / kV, v = i % kV;
    out[(size_t)b * kT * kV + v] = (v == 1) ? 1.0f : 0.0f;
}

// ============================================================================
extern "C" void kernel_launch(void* const* d_in, const int* in_sizes, int n_in,
                              void* d_out, int out_size, void* d_ws, size_t ws_size,
                              hipStream_t stream)
{
    (void)in_sizes; (void)n_in; (void)out_size; (void)ws_size;

    const float* enc    = (const float*)d_in[0];
    const int*   target = (const int*)  d_in[1];
    const float* emb    = (const float*)d_in[2];
    const float* Wq     = (const float*)d_in[3];
    const float* bq     = (const float*)d_in[4];
    const float* Wk     = (const float*)d_in[5];
    const float* bk     = (const float*)d_in[6];
    const float* Wa     = (const float*)d_in[7];
    const float* ba     = (const float*)d_in[8];
    const float* W_ih   = (const float*)d_in[9];
    const float* b_ih   = (const float*)d_in[10];
    const float* W_hh   = (const float*)d_in[11];
    const float* b_hh   = (const float*)d_in[12];
    const float* W_lin  = (const float*)d_in[13];
    const float* b_lin  = (const float*)d_in[14];
    float* out = (float*)d_out;

    // ---- carve workspace ----
    char* p = (char*)d_ws;
    auto carve = [&](size_t elems, size_t esz) {
        void* q = (void*)p;
        p += (elems * esz + 255) & ~(size_t)255;
        return q;
    };
    __bf16* enc_bf  = (__bf16*)carve((size_t)kB * kS * kH, 2);
    __bf16* emb_bf  = (__bf16*)carve((size_t)kV * kE, 2);
    __bf16* Wq_bf   = (__bf16*)carve((size_t)kH * kH, 2);
    __bf16* Wk_bf   = (__bf16*)carve((size_t)kH * kH, 2);
    __bf16* Wih_bf  = (__bf16*)carve((size_t)3 * kH * (kH + kE), 2);
    __bf16* Whh_bf  = (__bf16*)carve((size_t)3 * kH * kH, 2);
    __bf16* Wlin_bf = (__bf16*)carve((size_t)kV * kH, 2);
    float*  keys    = (float*) carve((size_t)kB * kS * kH, 4);
    float*  state   = (float*) carve((size_t)kB * kH, 4);
    __bf16* state_bf= (__bf16*)carve((size_t)kB * kH, 2);
    float*  qbuf    = (float*) carve((size_t)kB * kH, 4);
    float*  ctx     = (float*) carve((size_t)kB * kH, 4);
    __bf16* xbuf    = (__bf16*)carve((size_t)kB * (kE + kH), 2);
    float*  gi      = (float*) carve((size_t)kB * 3 * kH, 4);
    float*  gh      = (float*) carve((size_t)kB * 3 * kH, 4);

    auto cvt = [&](const float* src, __bf16* dst, size_t n) {
        cvt_bf16_kernel<<<dim3((unsigned)((n + 255) / 256)), dim3(256), 0, stream>>>(src, dst, n);
    };
    // M must be a multiple of 64 (register-blocked 4 m-tiles per wave), N multiple of 64.
    auto gemm = [&](const __bf16* A, int lda, const __bf16* W, int ldw, const float* bias,
                    float* C, long long ldc, int M, int N, int K) {
        dim3 grid((unsigned)(N / 64), (unsigned)(M / 64));
        wmma_gemm_kernel<<<grid, dim3(128), 0, stream>>>(A, lda, W, ldw, bias, C, ldc, K, N);
    };

    // ---- one-time per launch: bf16 conversions (weights become L2-resident) ----
    cvt(enc,   enc_bf,  (size_t)kB * kS * kH);
    cvt(emb,   emb_bf,  (size_t)kV * kE);
    cvt(Wq,    Wq_bf,   (size_t)kH * kH);
    cvt(Wk,    Wk_bf,   (size_t)kH * kH);
    cvt(W_ih,  Wih_bf,  (size_t)3 * kH * (kH + kE));
    cvt(W_hh,  Whh_bf,  (size_t)3 * kH * kH);
    cvt(W_lin, Wlin_bf, (size_t)kV * kH);

    // keys = encodings @ Wk^T + bk   (M = B*S = 8192, N = K = 1024)
    gemm(enc_bf, kH, Wk_bf, kH, bk, keys, kH, kB * kS, kH, kH);

    init_state_kernel<<<dim3((kB * kH + 255) / 256), dim3(256), 0, stream>>>(enc, state, state_bf);
    sos_kernel<<<dim3((kB * kV + 255) / 256), dim3(256), 0, stream>>>(out);

    // ---- 31 decode steps ----
    for (int t = 0; t < kT - 1; ++t) {
        // q = state @ Wq^T + bq
        gemm(state_bf, kH, Wq_bf, kH, bq, qbuf, kH, kB, kH, kH);
        // attention + softmax + context
        attn_kernel<<<dim3(kB), dim3(256), 0, stream>>>(qbuf, keys, enc, Wa, ba, ctx);
        // x = [embedding[target[:,t]], context]
        build_x_kernel<<<dim3((kB * (kE + kH)) / 256), dim3(256), 0, stream>>>(emb_bf, target, t, ctx, xbuf);
        // gi = x @ W_ih^T + b_ih ;  gh = state @ W_hh^T + b_hh
        gemm(xbuf, kE + kH, Wih_bf, kE + kH, b_ih, gi, 3 * kH, kB, 3 * kH, kE + kH);
        gemm(state_bf, kH, Whh_bf, kH, b_hh, gh, 3 * kH, kB, 3 * kH, kH);
        // GRU gates -> new state (in place)
        gru_gate_kernel<<<dim3((kB * kH + 255) / 256), dim3(256), 0, stream>>>(gi, gh, state, state_bf);
        // pred = new_state @ W_lin^T + b_lin  -> out[:, t+1, :]
        gemm(state_bf, kH, Wlin_bf, kH, b_lin, out + (size_t)(t + 1) * kV,
             (long long)kT * kV, kB, kV, kH);
    }
}
